// ExpertNet_56075093016665
// MI455X (gfx1250) — matrix operand order repaired and checked by reading, not compiled, not run on hardware
//
#include <hip/hip_runtime.h>

// Sizes from the reference
#define NROWS 32768
#define DIM   1024
#define HENC  512
#define NZ    64
#define KEXP  16
#define HEXP  256
#define NCLS  10

typedef __attribute__((ext_vector_type(16))) __bf16 v16bf;
typedef __attribute__((ext_vector_type(8)))  float  v8f;
typedef unsigned short u16;

#define WMMA_BF16(a, b, c) \
    __builtin_amdgcn_wmma_f32_16x16x32_bf16(false, (a), false, (b), (short)0, (c), false, false)

__device__ __forceinline__ u16 f2bf(float f) {
    union { float f; unsigned u; } x; x.f = f;
    unsigned u = x.u;
    u += 0x7fffu + ((u >> 16) & 1u);          // round to nearest even
    return (u16)(u >> 16);
}

__device__ __forceinline__ v16bf pack_frag(uint4 lo, uint4 hi) {
    union { uint4 q[2]; v16bf v; } x;
    x.q[0] = lo; x.q[1] = hi;
    return x.v;
}

// A-matrix 16x32 bf16 fragment: lane L (m = L&15, khalf = L>>4) holds
// K = khalf*8 + 0..7  and  K = 16 + khalf*8 + 0..7   (two 16B chunks)
__device__ __forceinline__ v16bf load_fragA(const u16* row, int khalf) {
    return pack_frag(*(const uint4*)(row + khalf * 8),
                     *(const uint4*)(row + 16 + khalf * 8));
}
// B-matrix 32x16 bf16 fragment from B^T storage (row n, contiguous k):
// lane L (n = L&15, khalf = L>>4) holds K = khalf*8 + 0..7 and K = 16 + khalf*8 + 0..7 (two 16B chunks)
__device__ __forceinline__ v16bf load_fragB(const u16* coltk, int khalf) {
    return pack_frag(*(const uint4*)(coltk + khalf * 8),
                     *(const uint4*)(coltk + 16 + khalf * 8));
}

// CDNA5 async global->LDS copy (ASYNCcnt-tracked, no VGPR data path).
// lds_addr = wave-relative LDS byte offset (low 32 bits of generic pointer).
__device__ __forceinline__ void async_ld_b128(unsigned lds_addr, const void* gaddr) {
    asm volatile("global_load_async_to_lds_b128 %0, %1, off"
                 :: "v"(lds_addr), "v"((unsigned long long)(size_t)gaddr)
                 : "memory");
}
__device__ __forceinline__ void wait_async0() {
    asm volatile("s_wait_asynccnt 0x0" ::: "memory");
}
__device__ __forceinline__ unsigned lds_off(const void* p) {
    return (unsigned)(size_t)p;
}

// ---------------------------------------------------------------------------
// Weight conversion: dst[b][c][r] = bf16(src[b][r][c])   (transpose to B^T)
// ---------------------------------------------------------------------------
__global__ __launch_bounds__(256) void transpose_bf16_kernel(
    const float* __restrict__ src, u16* __restrict__ dst, int rows, int cols) {
    int b = blockIdx.y;
    int i = blockIdx.x * 256 + threadIdx.x;
    int total = rows * cols;
    if (i < total) {
        int c = i / rows, r = i - c * rows;
        dst[(size_t)b * total + i] = f2bf(src[(size_t)b * total + (size_t)r * cols + c]);
    }
}

// W2 [K][HEXP][10] -> W2T bf16 [K][16][HEXP], zero-padded c >= 10
__global__ __launch_bounds__(256) void w2pad_kernel(
    const float* __restrict__ W2, u16* __restrict__ W2T) {
    int i = blockIdx.x * 256 + threadIdx.x;      // 16*16*256 = 65536
    int k = i >> 12;
    int c = (i >> 8) & 15;
    int h = i & 255;
    float v = (c < NCLS) ? W2[((size_t)k * HEXP + h) * NCLS + c] : 0.f;
    W2T[i] = f2bf(v);
}

// ---------------------------------------------------------------------------
// GEMM1: h = relu(X @ enc_W + enc_b)  [32768,1024]x[1024,512] -> bf16 h
// Double-buffered LDS; B tile via async-to-LDS, A tile (f32->bf16) via VGPRs,
// both prefetched for step k+1 while step k runs its 8 WMMAs per wave.
// ---------------------------------------------------------------------------
__global__ __launch_bounds__(256) void gemm1_kernel(
    const float* __restrict__ X, const u16* __restrict__ encWT,
    const float* __restrict__ enc_b, u16* __restrict__ h_b) {
    const int colblk = blockIdx.x;
    const int rowblk = blockIdx.y;
    const int t = threadIdx.x;
    const int w = t >> 5;
    const int lane = t & 31;
    const int lm = lane & 15;
    const int kh = lane >> 4;

    __shared__ u16 As[2][128 * 40];   // 128 rows x 32 k, stride 40 (80B)
    __shared__ u16 Bs[2][128 * 40];   // 128 cols x 32 k (B^T)

    const float* Xrow = X + (size_t)(rowblk * 128) * DIM;
    const u16*   Brow = encWT + (size_t)(colblk * 128) * DIM;

    // Per-thread staging coordinates (constant across k-steps)
    unsigned ldsB[2][2];
    size_t   gB[2];
    #pragma unroll
    for (int it = 0; it < 2; ++it) {
        int c = t + it * 256, col = c >> 2, seg = c & 3;
        gB[it] = (size_t)col * DIM + seg * 8;
        ldsB[0][it] = lds_off(&Bs[0][col * 40 + seg * 8]);
        ldsB[1][it] = lds_off(&Bs[1][col * 40 + seg * 8]);
    }
    u16*   pA[2][4];
    size_t gA[4];
    #pragma unroll
    for (int it = 0; it < 4; ++it) {
        int c = t + it * 256, row = c >> 3, seg = c & 7;
        gA[it] = (size_t)row * DIM + seg * 4;
        pA[0][it] = &As[0][row * 40 + seg * 4];
        pA[1][it] = &As[1][row * 40 + seg * 4];
    }

    // Prologue: stage k-step 0 into buffer 0
    #pragma unroll
    for (int it = 0; it < 2; ++it) async_ld_b128(ldsB[0][it], Brow + gB[it]);
    #pragma unroll
    for (int it = 0; it < 4; ++it) {
        float4 v = *(const float4*)(Xrow + gA[it]);
        u16* d = pA[0][it];
        d[0] = f2bf(v.x); d[1] = f2bf(v.y); d[2] = f2bf(v.z); d[3] = f2bf(v.w);
    }
    wait_async0();
    __syncthreads();

    v8f acc[8] = {};
    for (int ks = 0; ks < DIM; ks += 32) {
        const int  cur = (ks >> 5) & 1, nxt = cur ^ 1;
        const bool more = (ks + 32) < DIM;
        float4 xa[4];
        if (more) {
            #pragma unroll
            for (int it = 0; it < 2; ++it)
                async_ld_b128(ldsB[nxt][it], Brow + gB[it] + ks + 32);
            #pragma unroll
            for (int it = 0; it < 4; ++it)
                xa[it] = *(const float4*)(Xrow + gA[it] + ks + 32);
        }
        v16bf af = load_fragA(&As[cur][(w * 16 + lm) * 40], kh);
        #pragma unroll
        for (int tc = 0; tc < 8; ++tc) {
            v16bf bf = load_fragB(&Bs[cur][(tc * 16 + lm) * 40], kh);
            acc[tc] = WMMA_BF16(af, bf, acc[tc]);
        }
        if (more) {
            #pragma unroll
            for (int it = 0; it < 4; ++it) {
                u16* d = pA[nxt][it];
                d[0] = f2bf(xa[it].x); d[1] = f2bf(xa[it].y);
                d[2] = f2bf(xa[it].z); d[3] = f2bf(xa[it].w);
            }
        }
        wait_async0();
        __syncthreads();
    }
    // epilogue: bias + relu -> bf16
    #pragma unroll
    for (int tc = 0; tc < 8; ++tc) {
        int col = colblk * 128 + tc * 16 + lm;
        float bias = enc_b[col];
        #pragma unroll
        for (int r = 0; r < 8; ++r) {
            int m = rowblk * 128 + w * 16 + r + kh * 8;
            float v = acc[tc][r] + bias;
            h_b[(size_t)m * HENC + col] = f2bf(v > 0.f ? v : 0.f);
        }
    }
}

// ---------------------------------------------------------------------------
// GEMM2 + q: z = h @ z_W + z_b ; q = studentT(z, mu) normalized over K.
// Both tiles are bf16 -> fully async-to-LDS staged, double-buffered.
// ---------------------------------------------------------------------------
__global__ __launch_bounds__(256) void gemm2_q_kernel(
    const u16* __restrict__ h_b, const u16* __restrict__ zWT,
    const float* __restrict__ z_b, const float* __restrict__ mu,
    u16* __restrict__ z16, float* __restrict__ qout) {
    const int rowblk = blockIdx.x;
    const int t = threadIdx.x;
    const int w = t >> 5;
    const int lane = t & 31;
    const int lm = lane & 15;
    const int kh = lane >> 4;

    __shared__ u16   As[2][128 * 40];
    __shared__ u16   Bs[2][64 * 40];
    __shared__ float zs[128 * 68];

    const u16* hrow = h_b + (size_t)(rowblk * 128) * HENC;

    unsigned ldsA[2][2], ldsBB[2];
    size_t   gA[2], gB1;
    #pragma unroll
    for (int it = 0; it < 2; ++it) {
        int c = t + it * 256, row = c >> 2, seg = c & 3;
        gA[it] = (size_t)row * HENC + seg * 8;
        ldsA[0][it] = lds_off(&As[0][row * 40 + seg * 8]);
        ldsA[1][it] = lds_off(&As[1][row * 40 + seg * 8]);
    }
    {
        int col = t >> 2, seg = t & 3;
        gB1 = (size_t)col * HENC + seg * 8;
        ldsBB[0] = lds_off(&Bs[0][col * 40 + seg * 8]);
        ldsBB[1] = lds_off(&Bs[1][col * 40 + seg * 8]);
    }

    #pragma unroll
    for (int it = 0; it < 2; ++it) async_ld_b128(ldsA[0][it], hrow + gA[it]);
    async_ld_b128(ldsBB[0], zWT + gB1);
    wait_async0();
    __syncthreads();

    v8f acc[4] = {};
    for (int ks = 0; ks < HENC; ks += 32) {
        const int  cur = (ks >> 5) & 1, nxt = cur ^ 1;
        const bool more = (ks + 32) < HENC;
        if (more) {
            #pragma unroll
            for (int it = 0; it < 2; ++it)
                async_ld_b128(ldsA[nxt][it], hrow + gA[it] + ks + 32);
            async_ld_b128(ldsBB[nxt], zWT + gB1 + ks + 32);
        }
        v16bf af = load_fragA(&As[cur][(w * 16 + lm) * 40], kh);
        #pragma unroll
        for (int tc = 0; tc < 4; ++tc) {
            v16bf bf = load_fragB(&Bs[cur][(tc * 16 + lm) * 40], kh);
            acc[tc] = WMMA_BF16(af, bf, acc[tc]);
        }
        wait_async0();
        __syncthreads();
    }
    // bias, park z in LDS
    #pragma unroll
    for (int tc = 0; tc < 4; ++tc) {
        float bias = z_b[tc * 16 + lm];
        #pragma unroll
        for (int r = 0; r < 8; ++r)
            zs[(w * 16 + r + kh * 8) * 68 + tc * 16 + lm] = acc[tc][r] + bias;
    }
    __syncthreads();
    // Student-t soft assignment per row (threads 0..127)
    if (t < 128) {
        const float* zr = &zs[t * 68];
        float qv[KEXP], s = 0.f;
        #pragma unroll
        for (int k = 0; k < KEXP; ++k) {
            float d2 = 0.f;
            const float* muk = mu + k * NZ;
            #pragma unroll 8
            for (int d = 0; d < NZ; ++d) { float df = zr[d] - muk[d]; d2 += df * df; }
            qv[k] = 1.f / (1.f + d2);
            s += qv[k];
        }
        float inv = 1.f / s;
        #pragma unroll
        for (int k = 0; k < KEXP; ++k)
            qout[(size_t)(rowblk * 128 + t) * KEXP + k] = qv[k] * inv;
    }
    // z -> bf16 (32 contiguous elems per thread)
    #pragma unroll
    for (int it = 0; it < 32; ++it) {
        int e = t * 32 + it;
        int row = e >> 6, col = e & 63;
        z16[(size_t)(rowblk * 128 + row) * NZ + col] = f2bf(zs[row * 68 + col]);
    }
}

// ---------------------------------------------------------------------------
// Experts (fused): for k=0..15: eh = relu(z@W1[k]+b1[k]); logits = eh@W2[k]+b2[k];
// preds += q[:,k]*logits.  W1[k]/W2[k] async-prefetched into ping-pong LDS;
// eh re-fragments through a per-wave LDS buffer (intra-wave DS ordering).
// ---------------------------------------------------------------------------
__global__ __launch_bounds__(256) void expert_kernel(
    const u16* __restrict__ z16, const float* __restrict__ qg,
    const u16* __restrict__ W1T, const float* __restrict__ b1,
    const u16* __restrict__ W2T, const float* __restrict__ b2,
    float* __restrict__ out) {
    const int rowblk = blockIdx.x;
    const int t = threadIdx.x;
    const int w = t >> 5;
    const int lane = t & 31;
    const int lm = lane & 15;
    const int kh = lane >> 4;

    __shared__ float qs[128 * KEXP];        //  8 KB
    __shared__ u16   ehs[8][16 * 72];       // 18 KB, per-wave chunk
    __shared__ u16   w1s[2][HEXP * 72];     // 72 KB, W1[k]^T padded stride 72
    __shared__ u16   w2s[2][16 * 264];      // 16.5 KB, W2[k]^T padded stride 264

    // staging coordinates
    unsigned lw1[2][4], lw2[2][2];
    size_t   g1[4], g2[2];
    #pragma unroll
    for (int it = 0; it < 4; ++it) {
        int c = t + it * 256, n = c >> 2, seg = c & 3;
        g1[it] = (size_t)n * NZ + seg * 8;
        lw1[0][it] = lds_off(&w1s[0][n * 72 + seg * 8]);
        lw1[1][it] = lds_off(&w1s[1][n * 72 + seg * 8]);
    }
    #pragma unroll
    for (int it = 0; it < 2; ++it) {
        int c = t + it * 256, n = c >> 5, seg = c & 31;
        g2[it] = (size_t)n * HEXP + seg * 8;
        lw2[0][it] = lds_off(&w2s[0][n * 264 + seg * 8]);
        lw2[1][it] = lds_off(&w2s[1][n * 264 + seg * 8]);
    }

    // Prologue: q tile, z fragments, expert-0 weights
    #pragma unroll
    for (int it = 0; it < 8; ++it) {
        int e = t + it * 256;
        qs[e] = qg[(size_t)(rowblk * 128) * KEXP + e];
    }
    #pragma unroll
    for (int it = 0; it < 4; ++it) async_ld_b128(lw1[0][it], W1T + g1[it]);
    #pragma unroll
    for (int it = 0; it < 2; ++it) async_ld_b128(lw2[0][it], W2T + g2[it]);

    const u16* zp = z16 + (size_t)(rowblk * 128 + w * 16 + lm) * NZ;
    v16bf zA0 = pack_frag(*(const uint4*)(zp + kh * 8),      *(const uint4*)(zp + 16 + kh * 8));
    v16bf zA1 = pack_frag(*(const uint4*)(zp + 32 + kh * 8), *(const uint4*)(zp + 48 + kh * 8));

    wait_async0();
    __syncthreads();

    v8f pacc = {};
    u16* ew = &ehs[w][0];

    for (int k = 0; k < KEXP; ++k) {
        const int cur = k & 1, nxt = cur ^ 1;
        if (k + 1 < KEXP) {
            const u16* W1n = W1T + (size_t)(k + 1) * HEXP * NZ;
            const u16* W2n = W2T + (size_t)(k + 1) * 16 * HEXP;
            #pragma unroll
            for (int it = 0; it < 4; ++it) async_ld_b128(lw1[nxt][it], W1n + g1[it]);
            #pragma unroll
            for (int it = 0; it < 2; ++it) async_ld_b128(lw2[nxt][it], W2n + g2[it]);
        }
        v8f lacc = {};
        #pragma unroll
        for (int hc = 0; hc < 4; ++hc) {
            v8f eacc[4] = {};
            #pragma unroll
            for (int tc = 0; tc < 4; ++tc) {
                const u16* bp = &w1s[cur][(hc * 64 + tc * 16 + lm) * 72];
                eacc[tc] = WMMA_BF16(zA0, load_fragB(bp, kh), eacc[tc]);
                eacc[tc] = WMMA_BF16(zA1, load_fragB(bp + 32, kh), eacc[tc]);
            }
            // bias + relu -> per-wave LDS (intra-wave DS ordering, no barrier)
            #pragma unroll
            for (int tc = 0; tc < 4; ++tc) {
                float bias = b1[k * HEXP + hc * 64 + tc * 16 + lm];
                #pragma unroll
                for (int r = 0; r < 8; ++r) {
                    float v = eacc[tc][r] + bias;
                    ew[(r + kh * 8) * 72 + tc * 16 + lm] = f2bf(v > 0.f ? v : 0.f);
                }
            }
            // second GEMM partial: [16,64] x [64,16]
            #pragma unroll
            for (int ks = 0; ks < 2; ++ks) {
                v16bf af = pack_frag(*(const uint4*)(ew + lm * 72 + ks * 32 + kh * 8),
                                     *(const uint4*)(ew + lm * 72 + ks * 32 + 16 + kh * 8));
                v16bf bf = load_fragB(&w2s[cur][lm * 264 + hc * 64 + ks * 32], kh);
                lacc = WMMA_BF16(af, bf, lacc);
            }
        }
        float b2v = (lm < NCLS) ? b2[k * NCLS + lm] : 0.f;
        #pragma unroll
        for (int r = 0; r < 8; ++r) {
            float qv = qs[(w * 16 + r + kh * 8) * KEXP + k];
            pacc[r] += qv * (lacc[r] + b2v);
        }
        wait_async0();
        __syncthreads();
    }
    if (lm < NCLS) {
        #pragma unroll
        for (int r = 0; r < 8; ++r) {
            int m = rowblk * 128 + w * 16 + r + kh * 8;
            out[(size_t)m * NCLS + lm] = pacc[r];
        }
    }
}

// ---------------------------------------------------------------------------
extern "C" void kernel_launch(void* const* d_in, const int* in_sizes, int n_in,
                              void* d_out, int out_size, void* d_ws, size_t ws_size,
                              hipStream_t stream) {
    (void)in_sizes; (void)n_in; (void)out_size; (void)ws_size;
    const float* X     = (const float*)d_in[0];
    const float* enc_W = (const float*)d_in[1];
    const float* enc_b = (const float*)d_in[2];
    const float* z_W   = (const float*)d_in[3];
    const float* z_b   = (const float*)d_in[4];
    const float* mu    = (const float*)d_in[5];
    const float* W1    = (const float*)d_in[6];
    const float* b1    = (const float*)d_in[7];
    const float* W2    = (const float*)d_in[8];
    const float* b2    = (const float*)d_in[9];

    char* ws = (char*)d_ws;
    size_t off = 0;
    auto alloc = [&](size_t bytes) -> void* {
        void* p = ws + off;
        off = (off + bytes + 255) & ~(size_t)255;
        return p;
    };
    u16*   encWT = (u16*)alloc((size_t)HENC * DIM * 2);        // 1 MB
    u16*   zWT   = (u16*)alloc((size_t)NZ * HENC * 2);         // 64 KB
    u16*   W1Tb  = (u16*)alloc((size_t)KEXP * HEXP * NZ * 2);  // 512 KB
    u16*   W2Tb  = (u16*)alloc((size_t)KEXP * 16 * HEXP * 2);  // 128 KB
    u16*   h_b   = (u16*)alloc((size_t)NROWS * HENC * 2);      // 32 MB
    u16*   z16   = (u16*)alloc((size_t)NROWS * NZ * 2);        // 4 MB
    float* qbuf  = (float*)alloc((size_t)NROWS * KEXP * 4);    // 2 MB

    transpose_bf16_kernel<<<dim3((DIM * HENC + 255) / 256, 1), 256, 0, stream>>>(
        enc_W, encWT, DIM, HENC);
    transpose_bf16_kernel<<<dim3((HENC * NZ + 255) / 256, 1), 256, 0, stream>>>(
        z_W, zWT, HENC, NZ);
    transpose_bf16_kernel<<<dim3((NZ * HEXP + 255) / 256, KEXP), 256, 0, stream>>>(
        W1, W1Tb, NZ, HEXP);
    w2pad_kernel<<<(KEXP * 16 * HEXP) / 256, 256, 0, stream>>>(W2, W2Tb);

    gemm1_kernel<<<dim3(HENC / 128, NROWS / 128), 256, 0, stream>>>(X, encWT, enc_b, h_b);
    gemm2_q_kernel<<<NROWS / 128, 256, 0, stream>>>(h_b, zWT, z_b, mu, z16, qbuf);
    expert_kernel<<<NROWS / 128, 256, 0, stream>>>(z16, qbuf, W1Tb, b1, W2Tb, b2,
                                                   (float*)d_out);
}
